// PFNet3_30494267802296
// MI455X (gfx1250) — matrix operand order repaired and checked by reading, not compiled, not run on hardware
//
#include <hip/hip_runtime.h>
#include <hip/hip_bf16.h>
#include <math.h>

typedef __attribute__((ext_vector_type(16))) _Float16 v16h;
typedef __attribute__((ext_vector_type(8)))  float    v8f;

#define HDIM 32

// ---------------------------------------------------------------------------
// Input layer: h1 = leaky(x[N,3] @ w[3,32] + b), thread per (node, feature)
// ---------------------------------------------------------------------------
__global__ void k_in1(const float* __restrict__ x, const float* __restrict__ w,
                      const float* __restrict__ b, float* __restrict__ h1, int n) {
    int i = blockIdx.x * blockDim.x + threadIdx.x;
    int node = i >> 5, f = i & 31;
    if (node >= n) return;
    float v = b[f] + x[node*3+0]*w[0*HDIM+f] + x[node*3+1]*w[1*HDIM+f] + x[node*3+2]*w[2*HDIM+f];
    h1[i] = (v > 0.f) ? v : 0.01f * v;
}

// ---------------------------------------------------------------------------
// WMMA GEMM: C[N,32] = act(A[N,32] @ W[32,32] + bias)
// block = 128 threads = 4 waves; wave -> (rowtile = blk*2 + w>>1, coltile = w&1)
// ACT: 0 = none, 1 = leaky(0.01)
// SCALED: also write C2 = C * dis[row]^2  (GCN self-loop contribution)
// Fragment layouts per CDNA5 ISA 7.12.2 (wave32):
//   A 16x32 f16: lane l (M = l&15), g = l>>4; a[j] K = g*8+j (j<8), 16+g*8+j-8 (j>=8)
//   B 32x16 f16: lane l (N = l&15), g = l>>4; b[j] K = g*8+j (j<8), 16+g*8+j-8 (j>=8)
//   C 16x16 f32: lane l (N = l&15), g = l>>4; c[j] M = g*8+j
// ---------------------------------------------------------------------------
template<int ACT, bool SCALED>
__global__ void k_gemm32(const float* __restrict__ A, const float* __restrict__ W,
                         const float* __restrict__ bias, float* __restrict__ C,
                         const float* __restrict__ dis, float* __restrict__ C2, int nrows) {
    int wave = threadIdx.x >> 5;
    int lane = threadIdx.x & 31;
    int m0 = (blockIdx.x * 2 + (wave >> 1)) * 16;
    int n0 = (wave & 1) * 16;
    if (m0 >= nrows) return;
    int g = lane >> 4;
    int l = lane & 15;

    int arow = m0 + l;
    if (arow >= nrows) arow = nrows - 1;            // clamp loads for ragged tail
    const float* ap = A + (size_t)arow * HDIM;

    // A fragment: two 32B-aligned float4 pairs
    float4 x0 = *(const float4*)(ap + g*8);
    float4 x1 = *(const float4*)(ap + g*8 + 4);
    float4 x2 = *(const float4*)(ap + 16 + g*8);
    float4 x3 = *(const float4*)(ap + 16 + g*8 + 4);
    v16h a;
    a[0]=(_Float16)x0.x;  a[1]=(_Float16)x0.y;  a[2]=(_Float16)x0.z;  a[3]=(_Float16)x0.w;
    a[4]=(_Float16)x1.x;  a[5]=(_Float16)x1.y;  a[6]=(_Float16)x1.z;  a[7]=(_Float16)x1.w;
    a[8]=(_Float16)x2.x;  a[9]=(_Float16)x2.y;  a[10]=(_Float16)x2.z; a[11]=(_Float16)x2.w;
    a[12]=(_Float16)x3.x; a[13]=(_Float16)x3.y; a[14]=(_Float16)x3.z; a[15]=(_Float16)x3.w;

    v16h b;
    #pragma unroll
    for (int j = 0; j < 16; ++j) b[j] = (_Float16)W[(g*8 + j + (j & 8)) * HDIM + n0 + l];

    v8f c = {};
    c = __builtin_amdgcn_wmma_f32_16x16x32_f16(false, a, false, b, (short)0, c, false, false);

    float bn = bias ? bias[n0 + l] : 0.0f;
    int r0 = m0 + g*8;
    size_t base = (size_t)r0 * HDIM + n0 + l;
    if (m0 + 16 <= nrows) {
        // full tile: unconditional strided stores (the common path; wave-uniform branch)
        #pragma unroll
        for (int j = 0; j < 8; ++j) {
            float v = c[j] + bn;
            if (ACT == 1) v = (v > 0.f) ? v : 0.01f * v;
            C[base + (size_t)j * HDIM] = v;
            if (SCALED) {
                float d = dis[r0 + j];
                C2[base + (size_t)j * HDIM] = v * d * d;   // self-loop: norm = 1/deg
            }
        }
    } else {
        #pragma unroll
        for (int j = 0; j < 8; ++j) {
            if (r0 + j >= nrows) continue;
            float v = c[j] + bn;
            if (ACT == 1) v = (v > 0.f) ? v : 0.01f * v;
            C[base + (size_t)j * HDIM] = v;
            if (SCALED) {
                float d = dis[r0 + j];
                C2[base + (size_t)j * HDIM] = v * d * d;
            }
        }
    }
}

// ---------------------------------------------------------------------------
// BatchNorm statistics: per-feature sum / sumsq (LDS reduce, then atomics)
// ---------------------------------------------------------------------------
__global__ void k_zero(float* p, int n) {
    int i = blockIdx.x * blockDim.x + threadIdx.x;
    if (i < n) p[i] = 0.f;
}

__global__ void k_bn_stats(const float* __restrict__ h, float* __restrict__ sums, int n32) {
    __shared__ float s[64];
    if (threadIdx.x < 64) s[threadIdx.x] = 0.f;
    __syncthreads();
    int f = threadIdx.x & 31;
    float ls = 0.f, lq = 0.f;
    for (int i = blockIdx.x * blockDim.x + threadIdx.x; i < n32; i += gridDim.x * blockDim.x) {
        float v = h[i];
        ls += v; lq += v * v;
    }
    atomicAdd(&s[f], ls);
    atomicAdd(&s[32 + f], lq);
    __syncthreads();
    if (threadIdx.x < 64) atomicAdd(&sums[threadIdx.x], s[threadIdx.x]);
}

__global__ void k_bn_params(const float* __restrict__ sums, const float* __restrict__ gamma,
                            const float* __restrict__ beta, float* __restrict__ sc, float invN) {
    int f = threadIdx.x;  // 32 threads
    float mu  = sums[f] * invN;
    float var = sums[32 + f] * invN - mu * mu;
    float istd = rsqrtf(var + 1e-5f);
    float scale = gamma[f] * istd;
    sc[f] = scale;
    sc[32 + f] = beta[f] - mu * scale;
}

__global__ void k_bn_apply(float4* __restrict__ h, const float* __restrict__ sc, int n8) {
    int i = blockIdx.x * blockDim.x + threadIdx.x;
    if (i >= n8) return;
    int f0 = (i & 7) * 4;
    float4 v = h[i];
    v.x = v.x * sc[f0+0] + sc[32+f0+0];
    v.y = v.y * sc[f0+1] + sc[32+f0+1];
    v.z = v.z * sc[f0+2] + sc[32+f0+2];
    v.w = v.w * sc[f0+3] + sc[32+f0+3];
    h[i] = v;
}

// ---------------------------------------------------------------------------
// Degree (with self loop), dis = rsqrt(deg), per-edge symmetric norm
// ---------------------------------------------------------------------------
__global__ void k_fill1(float* p, int n) {
    int i = blockIdx.x * blockDim.x + threadIdx.x;
    if (i < n) p[i] = 1.0f;
}

__global__ void k_deg_count(const int* __restrict__ dst, float* __restrict__ deg, int e) {
    int i = blockIdx.x * blockDim.x + threadIdx.x;
    if (i < e) atomicAdd(&deg[dst[i]], 1.0f);
}

__global__ void k_rsqrt(const float* __restrict__ deg, float* __restrict__ dis, int n) {
    int i = blockIdx.x * blockDim.x + threadIdx.x;
    if (i < n) dis[i] = rsqrtf(deg[i]);
}

// wnorm[e] = dis[src[e]] * dis[dst[e]]   (reused by all 3 GCN layers)
__global__ void k_norm(const int* __restrict__ src, const int* __restrict__ dst,
                       const float* __restrict__ dis, float* __restrict__ wnorm, int e) {
    int i = blockIdx.x * blockDim.x + threadIdx.x;
    if (i < e) wnorm[i] = dis[src[i]] * dis[dst[i]];
}

// ---------------------------------------------------------------------------
// Edge scatter: each wave owns 32 edges. Lanes load 32 (src,dst,wnorm) triples
// coalesced, then loop broadcasting each edge via shfl; lane = feature gives a
// 128B-coalesced gather of hw[s][*] and same-cacheline f32 atomics into
// acc[d][*]. All tables are L2-resident (hw/acc = 12.8 MB each << 192 MB L2).
// ---------------------------------------------------------------------------
__global__ void k_edge(const int* __restrict__ src, const int* __restrict__ dst,
                       const float* __restrict__ wnorm, const float* __restrict__ hw,
                       float* __restrict__ acc, int e) {
    int lane = threadIdx.x & 31;
    int wid  = (blockIdx.x * blockDim.x + threadIdx.x) >> 5;
    int e0 = wid * 32;
    if (e0 >= e) return;
    int idx = e0 + lane;
    if (idx >= e) idx = e - 1;
    int   sv = src[idx];
    int   dv = dst[idx];
    float wv = wnorm[idx];
    int ne = e - e0; if (ne > 32) ne = 32;
    #pragma unroll 8
    for (int i = 0; i < ne; ++i) {
        int   s = __shfl(sv, i, 32);
        int   d = __shfl(dv, i, 32);
        float w = __shfl(wv, i, 32);
        atomicAdd(&acc[(size_t)d * HDIM + lane], w * hw[(size_t)s * HDIM + lane]);
    }
}

// ---------------------------------------------------------------------------
// out = elu(acc + bias), float4-vectorized
// ---------------------------------------------------------------------------
__device__ __forceinline__ float elu1(float v) {
    return (v > 0.f) ? v : (expf(v) - 1.f);
}

__global__ void k_bias_elu(const float4* __restrict__ acc, const float* __restrict__ b,
                           float4* __restrict__ out, int n8) {
    int i = blockIdx.x * blockDim.x + threadIdx.x;
    if (i >= n8) return;
    int f0 = (i & 7) * 4;
    float4 v = acc[i];
    v.x = elu1(v.x + b[f0+0]);
    v.y = elu1(v.y + b[f0+1]);
    v.z = elu1(v.z + b[f0+2]);
    v.w = elu1(v.w + b[f0+3]);
    out[i] = v;
}

// ---------------------------------------------------------------------------
// Output head: r3[N,4] = r2[N,32] @ wo3[32,4] + bo3 ; sigmoid on column 0
// ---------------------------------------------------------------------------
__global__ void k_tail(const float* __restrict__ r2, const float* __restrict__ wo3,
                       const float* __restrict__ bo3, float* __restrict__ out, int n) {
    int node = blockIdx.x * blockDim.x + threadIdx.x;
    if (node >= n) return;
    const float* row = r2 + (size_t)node * HDIM;
    float o0 = bo3[0], o1 = bo3[1], o2 = bo3[2], o3 = bo3[3];
    #pragma unroll
    for (int k = 0; k < HDIM; ++k) {
        float v = row[k];
        o0 += v * wo3[k*4+0];
        o1 += v * wo3[k*4+1];
        o2 += v * wo3[k*4+2];
        o3 += v * wo3[k*4+3];
    }
    o0 = 1.f / (1.f + expf(-o0));
    out[(size_t)node*4+0] = o0;
    out[(size_t)node*4+1] = o1;
    out[(size_t)node*4+2] = o2;
    out[(size_t)node*4+3] = o3;
}

// ---------------------------------------------------------------------------
extern "C" void kernel_launch(void* const* d_in, const int* in_sizes, int n_in,
                              void* d_out, int out_size, void* d_ws, size_t ws_size,
                              hipStream_t stream) {
    const float* x     = (const float*)d_in[0];
    const float* w_in1 = (const float*)d_in[1];
    const float* b_in1 = (const float*)d_in[2];
    const float* w_in2 = (const float*)d_in[3];
    const float* b_in2 = (const float*)d_in[4];
    const float* gamma = (const float*)d_in[5];
    const float* beta  = (const float*)d_in[6];
    const float* wg[3] = {(const float*)d_in[7],  (const float*)d_in[9],  (const float*)d_in[11]};
    const float* bg[3] = {(const float*)d_in[8],  (const float*)d_in[10], (const float*)d_in[12]};
    const float* wo1   = (const float*)d_in[13];
    const float* bo1   = (const float*)d_in[14];
    const float* wo2   = (const float*)d_in[15];
    const float* bo2   = (const float*)d_in[16];
    const float* wo3   = (const float*)d_in[17];
    const float* bo3   = (const float*)d_in[18];
    const int*   ei    = (const int*)d_in[19];

    const int N   = in_sizes[0] / 3;
    const int E   = in_sizes[19] / 2;
    const int N32 = N * HDIM;
    const int N8  = N * 8;                  // float4 count of [N,32]
    const int* src = ei;
    const int* dst = ei + E;

    // workspace layout (f32)
    float* ws    = (float*)d_ws;
    float* buf0  = ws;                      // h1 / hw / r1          [N*32]
    float* buf1  = ws + (size_t)N32;        // hcur (layer input)    [N*32]
    float* buf2  = ws + (size_t)2 * N32;    // acc / r2              [N*32]
    float* deg   = ws + (size_t)3 * N32;    // [N]
    float* dis   = deg + N;                 // [N]
    float* sums  = dis + N;                 // [64] sum / sumsq
    float* sc    = sums + 64;               // [64] scale / shift
    float* wnorm = sc + 64;                 // [E]

    const int BLK = 256;
    int gN32 = (N32 + BLK - 1) / BLK;
    int gN8  = (N8  + BLK - 1) / BLK;
    int gN   = (N   + BLK - 1) / BLK;
    int gE   = (E   + BLK - 1) / BLK;       // 1 edge/thread kernels AND shfl edge kernel
    int rowtiles = (N + 15) / 16;
    int gGemm = (rowtiles + 1) / 2;         // 2 row-tiles (x2 col-tiles) per 128-thr block

    // ---- input MLP ----
    k_in1<<<gN32, BLK, 0, stream>>>(x, w_in1, b_in1, buf0, N);
    k_gemm32<1, false><<<gGemm, 128, 0, stream>>>(buf0, w_in2, b_in2, buf1, nullptr, nullptr, N);

    // ---- BatchNorm (batch stats) ----
    k_zero<<<1, 64, 0, stream>>>(sums, 64);
    k_bn_stats<<<1024, BLK, 0, stream>>>(buf1, sums, N32);
    k_bn_params<<<1, 32, 0, stream>>>(sums, gamma, beta, sc, 1.0f / (float)N);
    k_bn_apply<<<gN8, BLK, 0, stream>>>((float4*)buf1, sc, N8);

    // ---- degrees / symmetric norm (reused across all 3 layers) ----
    k_fill1<<<gN, BLK, 0, stream>>>(deg, N);
    k_deg_count<<<gE, BLK, 0, stream>>>(dst, deg, E);
    k_rsqrt<<<gN, BLK, 0, stream>>>(deg, dis, N);
    k_norm<<<gE, BLK, 0, stream>>>(src, dst, dis, wnorm, E);

    // ---- 3x GCN layers ----
    for (int layer = 0; layer < 3; ++layer) {
        // hw = hcur @ Wg ; acc = hw * dis^2  (self-loop term, full overwrite of acc)
        k_gemm32<0, true><<<gGemm, 128, 0, stream>>>(buf1, wg[layer], nullptr, buf0, dis, buf2, N);
        // acc[dst] += wnorm[e] * hw[src]   (wave = 32 edges, lane = feature)
        k_edge<<<gE, BLK, 0, stream>>>(src, dst, wnorm, buf0, buf2, E);
        // hcur = elu(acc + bias)
        k_bias_elu<<<gN8, BLK, 0, stream>>>((const float4*)buf2, bg[layer], (float4*)buf1, N8);
    }

    // ---- output MLP ----
    k_gemm32<1, false><<<gGemm, 128, 0, stream>>>(buf1, wo1, bo1, buf0, nullptr, nullptr, N);
    k_gemm32<1, false><<<gGemm, 128, 0, stream>>>(buf0, wo2, bo2, buf2, nullptr, nullptr, N);
    k_tail<<<gN, BLK, 0, stream>>>(buf2, wo3, bo3, (float*)d_out, N);
}
